// CrossFeature_24240795419257
// MI455X (gfx1250) — matrix operand order repaired and checked by reading, not compiled, not run on hardware
//
#include <hip/hip_runtime.h>
#include <math.h>

// ---------------------------------------------------------------------------
// CrossFeature fused kernel for MI455X (gfx1250, wave32, WMMA bf16)
// out[b,n,e] = exp( sum_f x[b,f,e] * entmax15( (x[b]W^T)Q^T * E^-0.5 )[n,f] * value[n,f] )
// ---------------------------------------------------------------------------

typedef __attribute__((ext_vector_type(16))) __bf16 v16bf;
typedef __attribute__((ext_vector_type(8)))  __bf16 v8bf;
typedef __attribute__((ext_vector_type(8)))  float  v8f;

#define BATCH 2048
#define FDIM  64     // nfield
#define EDIM  256    // nemb
#define NQ    64     // ncross

// LDS strides (elements), padded to dodge bank conflicts, 16B aligned rows
#define XS   264     // [*][256] bf16 arrays  (row = 528 B)
#define XTS  72      // [*][64]  bf16 arrays  (row = 144 B)
#define GS   68      // [*][64]  f32 array    (row = 272 B)

#define SMEM_BYTES 131072  // Xbf 33792 + XbfT 36864 + Klds 33792 + G 17408 + AWbf 9216

// --- bf16 fragment loaders (CDNA5 ISA §7.12.2 layouts, wave32) -------------
// A 16x32: lane half g=0: k in {kb..kb+7} (V0-3), {kb+16..kb+23} (V4-7); g=1: +8
__device__ __forceinline__ v16bf load_fragA(const __bf16* row, int kb, int g) {
  v8bf lo = *(const v8bf*)(row + kb + g * 8);
  v8bf hi = *(const v8bf*)(row + kb + 16 + g * 8);
  return __builtin_shufflevector(lo, hi, 0,1,2,3,4,5,6,7,8,9,10,11,12,13,14,15);
}
// B 32x16: lane half g=0: k in {kb..kb+7}, {kb+16..kb+23}; g=1: +8 (same as A)
__device__ __forceinline__ v16bf load_fragB(const __bf16* row, int kb, int g) {
  v8bf lo = *(const v8bf*)(row + kb + g * 8);
  v8bf hi = *(const v8bf*)(row + kb + 16 + g * 8);
  return __builtin_shufflevector(lo, hi, 0,1,2,3,4,5,6,7,8,9,10,11,12,13,14,15);
}

// --- pre-pass: fp32 -> bf16 for the batch-invariant weights ----------------
__global__ void cvt_wq_kernel(const float* __restrict__ W, const float* __restrict__ Q,
                              __bf16* __restrict__ Wbf, __bf16* __restrict__ Qbf) {
  int i = blockIdx.x * blockDim.x + threadIdx.x;
  if (i < EDIM * EDIM) Wbf[i] = (__bf16)W[i];
  if (i < NQ * EDIM)   Qbf[i] = (__bf16)Q[i];
}

// --- fused per-batch kernel -------------------------------------------------
__global__ __launch_bounds__(256, 2)
void crossfeature_fused_kernel(const float* __restrict__ x,
                               const float* __restrict__ value,
                               const __bf16* __restrict__ Wbf,
                               const __bf16* __restrict__ Qbf,
                               float* __restrict__ out) {
  extern __shared__ char smem[];
  __bf16* Xbf  = (__bf16*)smem;                   // [64][XS]   x[b] as bf16 (row-major f,e)
  __bf16* XbfT = Xbf  + FDIM * XS;                // [256][XTS] x[b]^T       (row-major e,f)
  __bf16* Klds = XbfT + EDIM * XTS;               // [64][XS]   K = X W^T    (row-major f,d)
  float*  G    = (float*)(Klds + FDIM * XS);      // [64][GS]   att gate     (row-major n,f)
  __bf16* AWbf = (__bf16*)(G + NQ * GS);          // [64][XTS]  entmax*value (row-major n,f)

  const int tid  = threadIdx.x;
  const int lane = tid & 31;
  const int wave = tid >> 5;          // 8 waves
  const int ln   = lane & 15;
  const int g    = lane >> 4;
  const int b    = blockIdx.x;

  // ---- stage 0: stage x[b] into LDS (bf16, both orientations) ----
  const float* xb = x + (size_t)b * FDIM * EDIM;
  for (int i = tid; i < FDIM * EDIM; i += 256) {
    int f = i >> 8, e = i & 255;
    __bf16 v = (__bf16)xb[i];
    Xbf [f * XS  + e] = v;
    XbfT[e * XTS + f] = v;
  }
  __syncthreads();

  // ---- stage 1: K[f,d] = sum_e X[f,e] * W[d,e]  (M=f:4 tiles, N=d:16 tiles, K=e:256)
  for (int i = 0; i < 8; ++i) {
    int t = wave + 8 * i;
    int mt = t & 3, nt = t >> 2;
    v8f acc = {};
    const __bf16* arow = Xbf + (mt * 16 + ln) * XS;        // A: X row-major [f][e]
    const __bf16* brow = Wbf + (size_t)(nt * 16 + ln) * EDIM; // B: W row-major [d][e]
#pragma unroll
    for (int ks = 0; ks < 8; ++ks) {
      v16bf a  = load_fragA(arow, ks * 32, g);
      v16bf bb = load_fragB(brow, ks * 32, g);
      acc = __builtin_amdgcn_wmma_f32_16x16x32_bf16(false, a, false, bb,
                                                    (short)0, acc, false, false);
    }
    int d  = nt * 16 + ln;
    int fb = mt * 16 + 8 * g;
#pragma unroll
    for (int r = 0; r < 8; ++r)
      Klds[(fb + r) * XS + d] = (__bf16)acc[r];
  }
  __syncthreads();

  // ---- stage 2: G[n,f] = scale * sum_d Q[n,d] * K[f,d]  (M=n:4, N=f:4, K=d:256)
  const float scale = 0.0625f;  // 256^-0.5
  for (int i = 0; i < 2; ++i) {
    int t = wave + 8 * i;
    int mt = t & 3, nt = t >> 2;
    v8f acc = {};
    const __bf16* arow = Qbf  + (size_t)(mt * 16 + ln) * EDIM; // A: Q row-major [n][d]
    const __bf16* brow = Klds + (nt * 16 + ln) * XS;           // B: K row-major [f][d]
#pragma unroll
    for (int ks = 0; ks < 8; ++ks) {
      v16bf a  = load_fragA(arow, ks * 32, g);
      v16bf bb = load_fragB(brow, ks * 32, g);
      acc = __builtin_amdgcn_wmma_f32_16x16x32_bf16(false, a, false, bb,
                                                    (short)0, acc, false, false);
    }
    int fcol = nt * 16 + ln;
    int nb   = mt * 16 + 8 * g;
#pragma unroll
    for (int r = 0; r < 8; ++r)
      G[(nb + r) * GS + fcol] = acc[r] * scale;
  }
  __syncthreads();

  // ---- stage 3: entmax-1.5 over f (50 bisection iters), all 256 lanes busy
  // 4 lanes per row: lanes {r,r+8,r+16,r+24} of one wave share row = wave*8 + (lane&7)
  {
    const int row = wave * 8 + (lane & 7);
    const int q   = lane >> 3;  // quarter 0..3 -> cols q*16..q*16+15
    float Xs[16];
    const float* grow = G + row * GS + q * 16;
#pragma unroll
    for (int j = 0; j < 16; ++j) Xs[j] = 0.5f * grow[j];   // (alpha-1) * att

    float mx = -3.0e38f;
#pragma unroll
    for (int j = 0; j < 16; ++j) mx = fmaxf(mx, Xs[j]);
    mx = fmaxf(mx, __shfl_xor(mx, 8, 32));
    mx = fmaxf(mx, __shfl_xor(mx, 16, 32));

    float tau_lo = mx - 1.0f;
    float tau_hi = mx - 0.125f;                 // (1/64)^(alpha-1)
    float s = 0.0f;
#pragma unroll
    for (int j = 0; j < 16; ++j) { float z = fmaxf(Xs[j] - tau_lo, 0.0f); s += z * z; }
    s += __shfl_xor(s, 8, 32);
    s += __shfl_xor(s, 16, 32);
    float f_lo = s - 1.0f;
    float dm = tau_hi - tau_lo;

    for (int it = 0; it < 50; ++it) {
      dm *= 0.5f;
      float tau_m = tau_lo + dm;
      float fm = 0.0f;
#pragma unroll
      for (int j = 0; j < 16; ++j) { float z = fmaxf(Xs[j] - tau_m, 0.0f); fm += z * z; }
      fm += __shfl_xor(fm, 8, 32);
      fm += __shfl_xor(fm, 16, 32);
      fm -= 1.0f;
      if (fm * f_lo >= 0.0f) tau_lo = tau_m;
    }

    float p[16], ps = 0.0f;
#pragma unroll
    for (int j = 0; j < 16; ++j) { float z = fmaxf(Xs[j] - tau_lo, 0.0f); p[j] = z * z; ps += p[j]; }
    ps += __shfl_xor(ps, 8, 32);
    ps += __shfl_xor(ps, 16, 32);
    float inv = 1.0f / ps;
    const float* vrow = value + row * FDIM + q * 16;
#pragma unroll
    for (int j = 0; j < 16; ++j)
      AWbf[row * XTS + q * 16 + j] = (__bf16)(p[j] * inv * vrow[j]);
  }
  __syncthreads();

  // ---- stage 4: out[n,e] = exp( sum_f AW[n,f] * X[f,e] )  (M=n:4, N=e:16, K=f:64)
  float* outb = out + (size_t)b * NQ * EDIM;
  for (int i = 0; i < 8; ++i) {
    int t = wave + 8 * i;
    int mt = t & 3, nt = t >> 2;
    v8f acc = {};
    const __bf16* arow = AWbf + (mt * 16 + ln) * XTS;   // A: AW row-major [n][f]
    const __bf16* brow = XbfT + (nt * 16 + ln) * XTS;   // B: X^T row-major [e][f]
#pragma unroll
    for (int ks = 0; ks < 2; ++ks) {
      v16bf a  = load_fragA(arow, ks * 32, g);
      v16bf bb = load_fragB(brow, ks * 32, g);
      acc = __builtin_amdgcn_wmma_f32_16x16x32_bf16(false, a, false, bb,
                                                    (short)0, acc, false, false);
    }
    int e  = nt * 16 + ln;
    int nb = mt * 16 + 8 * g;
#pragma unroll
    for (int r = 0; r < 8; ++r)
      outb[(size_t)(nb + r) * EDIM + e] = expf(acc[r]);
  }
}

// ---------------------------------------------------------------------------
extern "C" void kernel_launch(void* const* d_in, const int* in_sizes, int n_in,
                              void* d_out, int out_size, void* d_ws, size_t ws_size,
                              hipStream_t stream) {
  (void)in_sizes; (void)n_in; (void)out_size; (void)ws_size;
  const float* x = (const float*)d_in[0];     // [2048,64,256]
  const float* W = (const float*)d_in[1];     // [256,256]
  const float* Q = (const float*)d_in[2];     // [64,256]
  const float* V = (const float*)d_in[3];     // [64,64]
  float* out = (float*)d_out;                 // [2048,64,256]

  __bf16* Wbf = (__bf16*)d_ws;                          // 256*256 bf16 = 128 KB
  __bf16* Qbf = (__bf16*)((char*)d_ws + EDIM * EDIM * sizeof(__bf16)); // +32 KB

  // allow 128 KB dynamic LDS per block (WGP has 320 KB)
  hipFuncSetAttribute((const void*)crossfeature_fused_kernel,
                      hipFuncAttributeMaxDynamicSharedMemorySize, SMEM_BYTES);

  cvt_wq_kernel<<<(EDIM * EDIM + 255) / 256, 256, 0, stream>>>(W, Q, Wbf, Qbf);
  crossfeature_fused_kernel<<<BATCH, 256, SMEM_BYTES, stream>>>(x, V, Wbf, Qbf, out);
}
